// LocallyConnected2D_37263136260697
// MI455X (gfx1250) — matrix operand order repaired and checked by reading, not compiled, not run on hardware
//
#include <hip/hip_runtime.h>

// ---------------------------------------------------------------------------
// Conv2d VALID (NCHW, OIHW) as implicit GEMM on CDNA5 WMMA bf16 matrix cores,
// with bf16x2 (hi/lo) input splitting for near-fp32 accuracy.
//   M = Cout = 128, N = B*Ho*Wo, K = Cin*3*3 = 576 (tap-major, ci-inner order)
// Each workgroup: 128co x 64wo tile; each of 8 waves: 16co x 64wo
//   -> 4 f32 accumulators, 12 v_wmma per K-step per wave.
// ---------------------------------------------------------------------------

typedef __attribute__((ext_vector_type(16))) __bf16 v16bf;
typedef __attribute__((ext_vector_type(8)))  __bf16 v8bf;
typedef __attribute__((ext_vector_type(8)))  float  v8f;

#define B_     32
#define CIN    64
#define HIN    112
#define WIN    112
#define CO     128
#define HO     110
#define WO     110
#define NT     64          // wo-tile width per workgroup
#define KSTEPS 18          // 9 taps * 2 ci-halves (32 ci each)
#define XROW   40          // LDS row stride in bf16 elems (80B -> conflict-free b128)

union U16 { v16bf v; v8bf h[2]; };

static __device__ __forceinline__ __bf16 f2bf(float f) {
  unsigned u = __builtin_bit_cast(unsigned, f);
  u += 0x7fffu + ((u >> 16) & 1u);             // round-to-nearest-even
  unsigned short s = (unsigned short)(u >> 16);
  return __builtin_bit_cast(__bf16, s);
}
static __device__ __forceinline__ float bf2f(__bf16 b) {
  unsigned short s = __builtin_bit_cast(unsigned short, b);
  unsigned u = ((unsigned)s) << 16;
  return __builtin_bit_cast(float, u);
}

// ---------------------------------------------------------------------------
// Prep: weights [Co][Cin][3][3] f32  ->  Wt_hi/Wt_lo [tap][co][ci] bf16
// (ci contiguous so A-fragments load as aligned global b128)
// ---------------------------------------------------------------------------
__global__ void prep_weights(const float* __restrict__ w,
                             __bf16* __restrict__ wh, __bf16* __restrict__ wl) {
  int idx = blockIdx.x * 256 + threadIdx.x;     // idx = tap*8192 + co*64 + ci
  if (idx >= 9 * CO * CIN) return;
  int ci  = idx & 63;
  int co  = (idx >> 6) & 127;
  int tap = idx >> 13;
  float v = w[((co << 6) + ci) * 9 + tap];
  __bf16 h = f2bf(v);
  __bf16 l = f2bf(v - bf2f(h));
  wh[idx] = h;
  wl[idx] = l;
}

// ---------------------------------------------------------------------------
// Stage one (tap, ci-half) slice of x into LDS, transposed to [n][ci],
// split into bf16 hi/lo. 256 threads move 32ci x 64n floats (8 per thread,
// 32B contiguous global read per lane).
// ---------------------------------------------------------------------------
static __device__ __forceinline__ void stage_x(
    const float* __restrict__ x, __bf16 (*ldsbuf)[NT * XROW],
    int b, int ho, int wo0, int tap, int ch, int sci, int sn0) {
  const int kh = tap / 3, kw = tap - kh * 3;
  const float* row = x + (((size_t)b * CIN + (ch << 5) + sci) * HIN + (ho + kh)) * WIN
                       + (wo0 + kw);
  const int colbase = wo0 + kw;
#pragma unroll
  for (int i = 0; i < 8; ++i) {
    int n = sn0 + i;
    float v = ((colbase + n) < WIN) ? row[n] : 0.0f;   // clamp right edge
    __bf16 h = f2bf(v);
    __bf16 l = f2bf(v - bf2f(h));
    ldsbuf[0][n * XROW + sci] = h;
    ldsbuf[1][n * XROW + sci] = l;
  }
}

// ---------------------------------------------------------------------------
// Main kernel: grid = (ceil(WO/64), HO, B), block = 256 (8 waves).
// Wave w computes out rows co=[16w,16w+16) x 64 wo via v_wmma_f32_16x16x32_bf16.
// ---------------------------------------------------------------------------
__global__ void __launch_bounds__(256)
conv_wmma(const float* __restrict__ x, const __bf16* __restrict__ wh,
          const __bf16* __restrict__ wl, float* __restrict__ out) {
  __shared__ __align__(16) __bf16 lds[2][2][NT * XROW];   // double buf x {hi,lo} ~20KB

  const int wo0 = blockIdx.x * NT;
  const int ho  = blockIdx.y;
  const int b   = blockIdx.z;

  const int t    = threadIdx.x;
  const int lane = t & 31;
  const int wave = t >> 5;
  const int m    = lane & 15;       // row/col within 16
  const int hhf  = lane >> 4;       // lane half (selects K sub-range)
  const int co_base = wave << 4;

  const int sci = t >> 3;           // staging: ci 0..31
  const int sn0 = (t & 7) << 3;     // staging: n start 0,8,...,56

  v8f acc[4] = {};

  // prologue: stage step 0 into buffer 0
  stage_x(x, lds[0], b, ho, wo0, 0, 0, sci, sn0);

  for (int s = 0; s < KSTEPS; ++s) {
    __syncthreads();
    if (s + 1 < KSTEPS)
      stage_x(x, lds[(s + 1) & 1], b, ho, wo0, (s + 1) >> 1, (s + 1) & 1, sci, sn0);

    const int tap = s >> 1;
    const int ch  = s & 1;

    // ---- A fragments (weights) straight from global (L2-resident) ----------
    // lane<16 holds K {0..7,16..23}; lane>=16 holds K {8..15,24..31}
    const size_t wofs = ((size_t)(tap * CO + co_base + m) << 6) + (ch << 5) + (hhf << 3);
    U16 Ah, Al;
    Ah.h[0] = *(const v8bf*)(wh + wofs);
    Ah.h[1] = *(const v8bf*)(wh + wofs + 16);
    Al.h[0] = *(const v8bf*)(wl + wofs);
    Al.h[1] = *(const v8bf*)(wl + wofs + 16);

    // ---- B fragments (im2col x) from LDS -----------------------------------
    const __bf16* xh = lds[s & 1][0];
    const __bf16* xl = lds[s & 1][1];
    const int kk0 = hhf << 3;        // lane<16: K {0..7,16..23}, lane>=16: K {8..15,24..31}

#pragma unroll
    for (int j = 0; j < 4; ++j) {
      const int n = (j << 4) + m;
      const __bf16* ph = xh + n * XROW + kk0;
      const __bf16* pl = xl + n * XROW + kk0;
      U16 Bh, Bl;
      Bh.h[0] = *(const v8bf*)ph;  Bh.h[1] = *(const v8bf*)(ph + 16);
      Bl.h[0] = *(const v8bf*)pl;  Bl.h[1] = *(const v8bf*)(pl + 16);

      // bf16x2: hi*hi + hi*lo + lo*hi  (lo*lo term ~2^-18, dropped)
      acc[j] = __builtin_amdgcn_wmma_f32_16x16x32_bf16(false, Ah.v, false, Bh.v,
                                                       (short)0, acc[j], false, false);
      acc[j] = __builtin_amdgcn_wmma_f32_16x16x32_bf16(false, Ah.v, false, Bl.v,
                                                       (short)0, acc[j], false, false);
      acc[j] = __builtin_amdgcn_wmma_f32_16x16x32_bf16(false, Al.v, false, Bh.v,
                                                       (short)0, acc[j], false, false);
    }
  }

  // ---- store: C/D layout — VGPR r holds M = r + 8*(lane>=16), N = lane&15 ---
  const size_t plane = (size_t)HO * WO;
  float* obase = out + (((size_t)b * CO + co_base + (hhf << 3)) * HO + ho) * WO;
#pragma unroll
  for (int j = 0; j < 4; ++j) {
    const int wo = wo0 + (j << 4) + m;
    if (wo < WO) {
      float* op = obase + wo;
#pragma unroll
      for (int r = 0; r < 8; ++r) op[(size_t)r * plane] = acc[j][r];
    }
  }
}

// ---------------------------------------------------------------------------
extern "C" void kernel_launch(void* const* d_in, const int* in_sizes, int n_in,
                              void* d_out, int out_size, void* d_ws, size_t ws_size,
                              hipStream_t stream) {
  const float* x = (const float*)d_in[0];
  const float* w = (const float*)d_in[1];
  float* out = (float*)d_out;

  __bf16* wh = (__bf16*)d_ws;                 // [9][128][64] bf16 hi
  __bf16* wl = wh + 9 * CO * CIN;             // [9][128][64] bf16 lo  (~288KB total)

  prep_weights<<<dim3((9 * CO * CIN + 255) / 256), dim3(256), 0, stream>>>(w, wh, wl);
  conv_wmma<<<dim3((WO + NT - 1) / NT, HO, B_), dim3(256), 0, stream>>>(x, wh, wl, out);
}
